// MixtureOfExperts_54202487275618
// MI455X (gfx1250) — matrix operand order repaired and checked by reading, compiled, not run on hardware
//
#include <hip/hip_runtime.h>
#include <hip/hip_bf16.h>
#include <stdint.h>

typedef __attribute__((ext_vector_type(16))) __bf16 v16bf;
typedef __attribute__((ext_vector_type(8)))  __bf16 v8bf;
typedef __attribute__((ext_vector_type(8)))  float  v8f;
typedef __attribute__((ext_vector_type(2)))  int    v2i;
typedef __attribute__((ext_vector_type(4)))  int    v4i;

#if __has_builtin(__builtin_amdgcn_global_load_async_to_lds_b128)
#define HAVE_ASYNC_LDS_B128 1
#endif
#if __has_builtin(__builtin_amdgcn_s_wait_asynccnt)
#define HAVE_WAIT_ASYNC 1
#endif

union BfFrag { v16bf v; v8bf h[2]; };

__device__ __forceinline__ v8f wmma_bf16(v16bf a, v16bf b, v8f c) {
  // 8 args: (neg_a, A, neg_b, B, c_mod, C, reuse_a, reuse_b)
  return __builtin_amdgcn_wmma_f32_16x16x32_bf16(false, a, false, b, (short)0, c,
                                                 false, false);
}

// ---------------------------------------------------------------------------
// Router: one wave per token. logits = x@rw + rb; top-2; softmax over the 2;
// scatter into dense combine[N,E] (zeros elsewhere).
// ---------------------------------------------------------------------------
__global__ void moe_router_kernel(const float* __restrict__ x,
                                  const float* __restrict__ rw,
                                  const float* __restrict__ rb,
                                  float* __restrict__ combine,
                                  int N, int D, int E) {
  const int lane = threadIdx.x & 31;
  const int wave = threadIdx.x >> 5;
  const int n = blockIdx.x * 8 + wave;
  if (n >= N) return;
  float acc[8];
#pragma unroll
  for (int e = 0; e < 8; ++e) acc[e] = 0.f;
  const float* xp = x + (size_t)n * D;
  for (int d = lane; d < D; d += 32) {
    const float xv = xp[d];
    const float* r = rw + (size_t)d * E;
#pragma unroll
    for (int e = 0; e < 8; ++e) acc[e] += xv * r[e];
  }
#pragma unroll
  for (int e = 0; e < 8; ++e)
    for (int m = 16; m >= 1; m >>= 1) acc[e] += __shfl_xor(acc[e], m, 32);
  if (lane == 0) {
    float lg[8];
#pragma unroll
    for (int e = 0; e < 8; ++e) lg[e] = acc[e] + rb[e];
    int i0 = 0;
#pragma unroll
    for (int e = 1; e < 8; ++e) if (lg[e] > lg[i0]) i0 = e;
    int i1 = (i0 == 0) ? 1 : 0;
#pragma unroll
    for (int e = 0; e < 8; ++e) if (e != i0 && lg[e] > lg[i1]) i1 = e;
    const float e1 = __expf(lg[i1] - lg[i0]);
    const float inv = 1.f / (1.f + e1);
    float* c = combine + (size_t)n * E;
#pragma unroll
    for (int e = 0; e < 8; ++e) c[e] = 0.f;
    c[i0] = inv;
    c[i1] = e1 * inv;
  }
}

// ---------------------------------------------------------------------------
// x (fp32, [N,D]) -> bf16, same layout. Vectorized x4.
// ---------------------------------------------------------------------------
struct Bf4 { __bf16 a, b, c, d; };
__global__ void cvt_x_bf16_kernel(const float* __restrict__ in,
                                  __bf16* __restrict__ out, long long n4) {
  const long long i = (long long)blockIdx.x * blockDim.x + threadIdx.x;
  if (i >= n4) return;
  const float4 f = ((const float4*)in)[i];
  Bf4 o;
  o.a = (__bf16)f.x; o.b = (__bf16)f.y; o.c = (__bf16)f.z; o.d = (__bf16)f.w;
  ((Bf4*)out)[i] = o;
}

// ---------------------------------------------------------------------------
// W/V fp32 [E,D,H] -> bf16 [E,H,D] (transposed so WMMA B-fragments are
// K-contiguous). 32x32 tiles via padded LDS. grid.z selects (expert, W-or-V).
// ---------------------------------------------------------------------------
__global__ void transpose_cvt_kernel(const float* __restrict__ Wsrc,
                                     const float* __restrict__ Vsrc,
                                     __bf16* __restrict__ Wt,
                                     __bf16* __restrict__ Vt,
                                     int D, int H) {
  __shared__ float tile[32][33];
  const int e = blockIdx.z >> 1;
  const float* src = (blockIdx.z & 1) ? Vsrc : Wsrc;
  __bf16* dst = (blockIdx.z & 1) ? Vt : Wt;
  const int h0 = blockIdx.x * 32, d0 = blockIdx.y * 32;
  const int tx = threadIdx.x, ty = threadIdx.y;  // 32 x 8
  const float* s = src + (size_t)e * D * H;
#pragma unroll
  for (int i = 0; i < 32; i += 8)
    tile[ty + i][tx] = s[(size_t)(d0 + ty + i) * H + (h0 + tx)];
  __syncthreads();
  __bf16* o = dst + (size_t)e * H * D;
#pragma unroll
  for (int i = 0; i < 32; i += 8)
    o[(size_t)(h0 + ty + i) * D + (d0 + tx)] = (__bf16)tile[tx][ty + i];
}

// ---------------------------------------------------------------------------
// Main dense-MoE WMMA kernel, v2 (higher arithmetic intensity vs L2).
//
// Block = 256 threads = 8 waves as 2(M) x 4(N). Block tile = 128 tokens x
// 64 H-cols. Expert loop is OUTER; per expert the wave keeps 4 M-subtile
// accumulator pairs (a and b), so each B fragment fetched from the
// L2-resident bf16 weights feeds 4 WMMAs (8 per bw+bv pair): ~64 FLOP per
// L2 byte on the weight side. The 128x32 x tile is staged to LDS with
// async-to-LDS b128 copies and read back as A fragments via ds_load_b128.
// Epilogue per expert applies SwiGLU and the router combine weight into a
// persistent fp32 accumulator; final store writes out once.
// ---------------------------------------------------------------------------
__global__ __launch_bounds__(256, 1) void moe_gemm_kernel(
    const __bf16* __restrict__ xb, const __bf16* __restrict__ Wt,
    const __bf16* __restrict__ Vt, const float* __restrict__ combine,
    float* __restrict__ out, int N, int D, int H, int E) {
  __shared__ __attribute__((aligned(16))) __bf16 xs[128 * 32];

  const int tid = threadIdx.x;
  const int lane = tid & 31;
  const int wave = tid >> 5;
  const int waveM = wave >> 2;      // 0..1 -> which 64-token half
  const int waveN = wave & 3;       // 0..3 -> 16-col group
  const int tokBase = blockIdx.y * 128;
  const int col = blockIdx.x * 64 + waveN * 16 + (lane & 15);
  const int khalf = lane >> 4;      // 0 or 1
  const int mbase = waveM * 64;

  // Persistent output accumulator: 4 subtiles x 8 rows.
  float outv[4][8];
#pragma unroll
  for (int s = 0; s < 4; ++s)
#pragma unroll
    for (int i = 0; i < 8; ++i) outv[s][i] = 0.f;

  // LDS staging: thread copies 32B (two b128) of the 128x32 x tile.
  const int sRow = tid >> 1;
  const int sSeg = (tid & 1) * 16;
  const __bf16* gsrcBase = xb + (size_t)(tokBase + sRow) * D + sSeg;
  __bf16* ldst = &xs[sRow * 32 + sSeg];

  // A fragment sources (this code's contiguous k = 16*khalf + s, same order as B): row r, K contiguous at
  // r*32 + khalf*16 and +8.
  int aOff[4];
#pragma unroll
  for (int s = 0; s < 4; ++s)
    aOff[s] = (mbase + s * 16 + (lane & 15)) * 32 + khalf * 16;

  const v8f vzero = {0.f, 0.f, 0.f, 0.f, 0.f, 0.f, 0.f, 0.f};

  for (int e = 0; e < E; ++e) {
    v8f accA[4], accB[4];
#pragma unroll
    for (int s = 0; s < 4; ++s) { accA[s] = vzero; accB[s] = vzero; }

    const size_t boff = ((size_t)e * H + col) * (size_t)D + (size_t)khalf * 16;
    const __bf16* wbase = Wt + boff;
    const __bf16* vbase = Vt + boff;

    for (int k0 = 0; k0 < D; k0 += 32) {
      __syncthreads();  // previous tile fully consumed before overwrite
#ifdef HAVE_ASYNC_LDS_B128
      __builtin_amdgcn_global_load_async_to_lds_b128(
          (__attribute__((address_space(1))) v4i*)(gsrcBase + k0),
          (__attribute__((address_space(3))) v4i*)ldst, 0, 0);
      __builtin_amdgcn_global_load_async_to_lds_b128(
          (__attribute__((address_space(1))) v4i*)(gsrcBase + k0 + 8),
          (__attribute__((address_space(3))) v4i*)(ldst + 8), 0, 0);
#ifdef HAVE_WAIT_ASYNC
      __builtin_amdgcn_s_wait_asynccnt(0);
#endif
#else
      *(uint4*)ldst = *(const uint4*)(gsrcBase + k0);
      *(uint4*)(ldst + 8) = *(const uint4*)(gsrcBase + k0 + 8);
#endif
      if (k0 + 32 < D) __builtin_prefetch(gsrcBase + k0 + 32, 0, 1);
      __syncthreads();

      // B fragments (shared by 4 M-subtiles each).
      BfFrag bw, bv;
      const __bf16* wp = wbase + k0;
      bw.h[0] = *(const v8bf*)wp;
      bw.h[1] = *(const v8bf*)(wp + 8);
      const __bf16* vp = vbase + k0;
      bv.h[0] = *(const v8bf*)vp;
      bv.h[1] = *(const v8bf*)(vp + 8);

      BfFrag af[4];
#pragma unroll
      for (int s = 0; s < 4; ++s) {
        af[s].h[0] = *(const v8bf*)&xs[aOff[s]];
        af[s].h[1] = *(const v8bf*)&xs[aOff[s] + 8];
      }
#pragma unroll
      for (int s = 0; s < 4; ++s) accA[s] = wmma_bf16(af[s].v, bw.v, accA[s]);
#pragma unroll
      for (int s = 0; s < 4; ++s) accB[s] = wmma_bf16(af[s].v, bv.v, accB[s]);
    }

    // Per-expert epilogue: y = a*b*sigmoid(b), weighted by combine[row,e].
    // C/D layout: VGPR i -> row i (lanes 0-15) or 8+i (lanes 16-31).
#pragma unroll
    for (int s = 0; s < 4; ++s) {
      const int r0 = tokBase + mbase + s * 16 + (khalf << 3);
#pragma unroll
      for (int i = 0; i < 8; ++i) {
        const float b = accB[s][i];
        const float sig = 1.f / (1.f + __expf(-b));
        const float y = accA[s][i] * b * sig;
        outv[s][i] += combine[(size_t)(r0 + i) * E + e] * y;
      }
    }
  }

#pragma unroll
  for (int s = 0; s < 4; ++s) {
    const int r0 = tokBase + mbase + s * 16 + (khalf << 3);
#pragma unroll
    for (int i = 0; i < 8; ++i)
      out[(size_t)(r0 + i) * H + col] = outv[s][i];
  }
}

// ---------------------------------------------------------------------------
// Launcher
// ---------------------------------------------------------------------------
extern "C" void kernel_launch(void* const* d_in, const int* in_sizes, int n_in,
                              void* d_out, int out_size, void* d_ws,
                              size_t ws_size, hipStream_t stream) {
  const float* x  = (const float*)d_in[0];
  const float* W  = (const float*)d_in[1];
  const float* V  = (const float*)d_in[2];
  const float* rw = (const float*)d_in[3];
  const float* rb = (const float*)d_in[4];
  // d_in[5] = top_k (fixed at 2 per the reference setup; router hardcodes 2).

  const int E = in_sizes[4];                                  // 8
  const int D = in_sizes[3] / E;                              // 2048
  const int N = (int)((long long)in_sizes[0] / D);            // 4096
  const int H = (int)((long long)in_sizes[1] / ((long long)E * D));  // 2048

  char* ws = (char*)d_ws;
  size_t off = 0;
  auto alloc = [&](size_t bytes) {
    void* p = (void*)(ws + off);
    off = (off + bytes + 255) & ~(size_t)255;
    return p;
  };
  float*  combine = (float*)alloc((size_t)N * E * sizeof(float));
  __bf16* xb      = (__bf16*)alloc((size_t)N * D * sizeof(__bf16));
  __bf16* Wt      = (__bf16*)alloc((size_t)E * H * D * sizeof(__bf16));
  __bf16* Vt      = (__bf16*)alloc((size_t)E * H * D * sizeof(__bf16));
  (void)ws_size;

  moe_router_kernel<<<dim3((N + 7) / 8), 256, 0, stream>>>(x, rw, rb, combine,
                                                           N, D, E);

  const long long n4 = (long long)N * D / 4;
  cvt_x_bf16_kernel<<<dim3((unsigned)((n4 + 255) / 256)), 256, 0, stream>>>(
      x, xb, n4);

  transpose_cvt_kernel<<<dim3(H / 32, D / 32, E * 2), dim3(32, 8), 0, stream>>>(
      W, V, Wt, Vt, D, H);

  moe_gemm_kernel<<<dim3(H / 64, N / 128), 256, 0, stream>>>(
      xb, Wt, Vt, combine, (float*)d_out, N, D, H, E);
}
